// RPN_2972117368947
// MI455X (gfx1250) — matrix operand order repaired and checked by reading, not compiled, not run on hardware
//
#include <hip/hip_runtime.h>
#include <hip/hip_bf16.h>

typedef __attribute__((ext_vector_type(16))) _Float16 v16h;
typedef __attribute__((ext_vector_type(8)))  _Float16 v8h;
typedef __attribute__((ext_vector_type(8)))  float    v8f;

// Problem constants (from reference): B=4, H=W=256, Cin=64, Cmid=512, A=9.
#define BATCH 4
#define HH 256
#define WW 256
#define CIN 64
#define CMID 512
#define NA 9
#define NOUT 45            // 9 scores + 36 deltas
#define NPAD 48            // padded to 3 WMMA n-tiles
#define KBASE (3*3*CIN)    // 576 = 18 k-steps of 32
#define KS_A (KBASE/32)    // 18
#define KS_B (CMID/32)     // 16
#define TM 64              // spatial positions per workgroup
#define NT_A (CMID/16)     // 32 n-tiles for trunk
#define NT_B (NPAD/16)     // 3 n-tiles for heads

// d_ws layout (f16): [KS_A][NT_A][32 lanes][16 halfs] then [KS_B][NT_B][32][16]
#define WB_FRAG_ELEMS (KS_A*NT_A*32*16)   // 294912
#define WC_FRAG_ELEMS (KS_B*NT_B*32*16)   // 24576

// ---------------------------------------------------------------------------
// Prepack: fp32 weights -> f16 B-matrix fragments in the kernel's contiguous-K (A-matching)
// layout: lane 0-15 -> N=lane, K in [0,16); lane 16-31 -> same N, K in [16,32);
// half i of the 16-half fragment holds K = (lane>>4)*16 + i.
// ---------------------------------------------------------------------------
__global__ void rpn_prepack(const float* __restrict__ wb,
                            const float* __restrict__ wcls,
                            const float* __restrict__ wreg,
                            _Float16* __restrict__ wbfrag,
                            _Float16* __restrict__ wcfrag) {
  int idx = blockIdx.x * 256 + threadIdx.x;
  if (idx < WB_FRAG_ELEMS) {
    int i    = idx & 15;
    int lane = (idx >> 4) & 31;
    int t    = (idx >> 9) & 31;       // n-tile
    int s    = idx >> 14;             // k-step
    int k = s * 32 + ((lane >> 4) << 4) + i;
    int n = t * 16 + (lane & 15);
    wbfrag[idx] = (_Float16)wb[k * CMID + n];   // HWIO flattened: [k][n]
  } else {
    int idx2 = idx - WB_FRAG_ELEMS;
    if (idx2 < WC_FRAG_ELEMS) {
      int i    = idx2 & 15;
      int lane = (idx2 >> 4) & 31;
      int rem  = idx2 >> 9;
      int t    = rem % NT_B;
      int s    = rem / NT_B;
      int k = s * 32 + ((lane >> 4) << 4) + i;
      int n = t * 16 + (lane & 15);
      float w = 0.f;
      if (n < NA)        w = wcls[k * NA + n];
      else if (n < NOUT) w = wreg[k * (4 * NA) + (n - NA)];
      wcfrag[idx2] = (_Float16)w;
    }
  }
}

// ---------------------------------------------------------------------------
// Fused RPN: 3x3 conv (implicit GEMM, WMMA f16->f32) + ReLU kept in LDS,
// then 1x1 cls/reg heads (WMMA over LDS feat), sigmoid + objectness mask.
// Workgroup = 64 positions of one (b,h) row; 8 wave32s.
// ---------------------------------------------------------------------------
__global__ __launch_bounds__(256)
void rpn_main(const float* __restrict__ x,
              const float* __restrict__ b_base,
              const float* __restrict__ b_cls,
              const float* __restrict__ b_reg,
              const _Float16* __restrict__ wbfrag,
              const _Float16* __restrict__ wcfrag,
              float* __restrict__ out) {
  extern __shared__ __align__(16) char smem_raw[];
  _Float16* xs   = (_Float16*)smem_raw;                 // [3][66][64] f16 halo tile
  _Float16* feat = xs + 3 * 66 * CIN;                   // [64][512]  f16 trunk output
  float*    o2   = (float*)(feat + TM * CMID);          // [64][48]   f32 head output

  const int wg   = blockIdx.x;          // 0..4095
  const int xblk = wg & 3;
  const int row  = wg >> 2;             // b*H + h
  const int h    = row & (HH - 1);
  const int b    = row >> 8;
  const int x0   = xblk * TM;

  const int tid  = threadIdx.x;
  const int lane = tid & 31;
  const int wave = tid >> 5;
  const int mrow = lane & 15;           // M within tile
  const int khalf = lane >> 4;          // 0/1: selects K sub-range of A frag

  // ---- Stage 0: stage x halo (3 rows x 66 cols x 64 ch) into LDS as f16 ----
  for (int g = tid; g < 3 * 66 * (CIN / 4); g += 256) {
    int c4 = g & 15;
    int q  = (g >> 4) % 66;
    int r  = g / (66 * 16);
    int gy = h - 1 + r;
    int gx = x0 - 1 + q;
    float4 v = make_float4(0.f, 0.f, 0.f, 0.f);
    if ((unsigned)gy < (unsigned)HH && (unsigned)gx < (unsigned)WW)
      v = *(const float4*)(x + (((b * HH + gy) * WW + gx) * CIN + c4 * 4));
    _Float16* d = xs + ((r * 66 + q) * CIN + c4 * 4);
    d[0] = (_Float16)v.x; d[1] = (_Float16)v.y;
    d[2] = (_Float16)v.z; d[3] = (_Float16)v.w;
  }
  __syncthreads();

  // ---- Stage A: trunk GEMM  M=64, N=512 (this wave: 64 ch), K=576 ----
  v8f acc[4][4];
  #pragma unroll
  for (int mt = 0; mt < 4; ++mt)
    #pragma unroll
    for (int nt = 0; nt < 4; ++nt)
      acc[mt][nt] = (v8f)(0.f);

  for (int s = 0; s < KS_A; ++s) {
    const int ky = s / 6;
    const int kx = (s >> 1) % 3;
    const int c0 = (s & 1) * 32;
    v16h bf[4];
    #pragma unroll
    for (int nt = 0; nt < 4; ++nt) {
      int t = wave * 4 + nt;
      bf[nt] = *(const v16h*)(wbfrag + (((s * NT_A + t) * 32) + lane) * 16);
    }
    #pragma unroll
    for (int mt = 0; mt < 4; ++mt) {
      const int p = mt * 16 + mrow;
      const _Float16* ab = xs + ((ky * 66 + p + kx) * CIN + c0 + khalf * 16);
      union { v16h v; v8h h[2]; } au;
      au.h[0] = *(const v8h*)ab;          // K = kbase..kbase+7
      au.h[1] = *(const v8h*)(ab + 8);    // K = kbase+8..kbase+15
      #pragma unroll
      for (int nt = 0; nt < 4; ++nt)
        acc[mt][nt] = __builtin_amdgcn_wmma_f32_16x16x32_f16(
            false, au.v, false, bf[nt], (short)0, acc[mt][nt], false, false);
    }
  }

  // bias + ReLU -> f16 feat in LDS
  #pragma unroll
  for (int nt = 0; nt < 4; ++nt) {
    const int col = wave * 64 + nt * 16 + mrow;
    const float bb = b_base[col];
    #pragma unroll
    for (int mt = 0; mt < 4; ++mt) {
      #pragma unroll
      for (int v = 0; v < 8; ++v) {
        const int rm = mt * 16 + v + (khalf << 3);
        feat[rm * CMID + col] = (_Float16)fmaxf(acc[mt][nt][v] + bb, 0.f);
      }
    }
  }
  __syncthreads();

  // ---- Stage B: heads GEMM  M=64, N=48, K=512 over LDS feat ----
  for (int job = wave; job < 4 * NT_B; job += 8) {     // wave-uniform bound
    const int mt = job / NT_B;
    const int nt = job % NT_B;
    v8f c = (v8f)(0.f);
    for (int s = 0; s < KS_B; ++s) {
      const _Float16* ab = feat + (mt * 16 + mrow) * CMID + s * 32 + khalf * 16;
      union { v16h v; v8h h[2]; } au;
      au.h[0] = *(const v8h*)ab;
      au.h[1] = *(const v8h*)(ab + 8);
      v16h bf = *(const v16h*)(wcfrag + (((s * NT_B + nt) * 32) + lane) * 16);
      c = __builtin_amdgcn_wmma_f32_16x16x32_f16(
          false, au.v, false, bf, (short)0, c, false, false);
    }
    #pragma unroll
    for (int v = 0; v < 8; ++v) {
      const int rm = mt * 16 + v + (khalf << 3);
      o2[rm * NPAD + nt * 16 + mrow] = c[v];
    }
  }
  __syncthreads();

  // ---- Epilogue: sigmoid scores, objectness-masked deltas ----
  for (int e = tid; e < TM * NOUT; e += 256) {
    const int p = e / NOUT;
    const int j = e % NOUT;
    float val;
    if (j < NA) {
      val = 1.f / (1.f + __expf(-(o2[p * NPAD + j] + b_cls[j])));
    } else {
      const int d = j - NA;
      const int a = d >> 2;
      const float sc = 1.f / (1.f + __expf(-(o2[p * NPAD + a] + b_cls[a])));
      val = (sc > 0.7f) ? (o2[p * NPAD + j] + b_reg[d]) : 0.f;
    }
    out[(row * WW + x0 + p) * NOUT + j] = val;
  }
}

// ---------------------------------------------------------------------------
extern "C" void kernel_launch(void* const* d_in, const int* in_sizes, int n_in,
                              void* d_out, int out_size, void* d_ws, size_t ws_size,
                              hipStream_t stream) {
  (void)in_sizes; (void)n_in; (void)out_size; (void)ws_size;
  const float* x      = (const float*)d_in[0];
  const float* w_base = (const float*)d_in[1];
  const float* b_base = (const float*)d_in[2];
  const float* w_cls  = (const float*)d_in[3];
  const float* b_cls  = (const float*)d_in[4];
  const float* w_reg  = (const float*)d_in[5];
  const float* b_reg  = (const float*)d_in[6];
  (void)b_reg;

  _Float16* wbfrag = (_Float16*)d_ws;
  _Float16* wcfrag = wbfrag + WB_FRAG_ELEMS;

  const int prep_elems = WB_FRAG_ELEMS + WC_FRAG_ELEMS;
  rpn_prepack<<<(prep_elems + 255) / 256, 256, 0, stream>>>(
      w_base, w_cls, w_reg, wbfrag, wcfrag);

  const size_t shmem = (3 * 66 * CIN + TM * CMID) * sizeof(_Float16)
                     + TM * NPAD * sizeof(float);   // ~101 KB (<320 KB WGP LDS)
  const int ngrid = (BATCH * HH * WW) / TM;         // 4096
  rpn_main<<<ngrid, 256, shmem, stream>>>(
      x, b_base, (const float*)d_in[4], (const float*)d_in[6],
      wbfrag, wcfrag, (float*)d_out);
  (void)b_cls;
}
